// ResidualBlock_4389456576692
// MI455X (gfx1250) — matrix operand order repaired and checked by reading, not compiled, not run on hardware
//
#include <hip/hip_runtime.h>

typedef __attribute__((ext_vector_type(16))) __bf16 v16bf;
typedef __attribute__((ext_vector_type(8)))  float  v8f;

#define CCH    64
#define KOFF   27
#define TILE_M 128
#define LEAK   (1.0f / 3.0f)
#define BN_EPS 1e-4f

__device__ __forceinline__ unsigned short f2bf(float f) {
  unsigned int u = __float_as_uint(f);
  u += 0x7FFFu + ((u >> 16) & 1u);   // round-to-nearest-even
  return (unsigned short)(u >> 16);
}

__device__ __forceinline__ unsigned int pack2bf(float a, float b) {
  return (unsigned int)f2bf(a) | ((unsigned int)f2bf(b) << 16);
}

// ---- CDNA5 async copy: global -> LDS, 128-bit, ASYNCcnt-tracked ----
__device__ __forceinline__ void async_ld128(unsigned int lds_off,
                                            const unsigned int* gsrc) {
  asm volatile("global_load_async_to_lds_b128 %0, %1, off"
               :: "v"(lds_off), "v"(gsrc)
               : "memory");
}

__device__ __forceinline__ void wait_async0() {
#if __has_builtin(__builtin_amdgcn_s_wait_asynccnt)
  __builtin_amdgcn_s_wait_asynccnt(0);
#else
  asm volatile("s_wait_asynccnt 0x0" ::: "memory");
#endif
}

// ---------------- fp32 -> bf16 pair cast (features) ----------------
__global__ void __launch_bounds__(256)
cast_pairs_kernel(const float2* __restrict__ src, unsigned int* __restrict__ dst,
                  int npairs) {
  int i = blockIdx.x * blockDim.x + threadIdx.x;
  if (i < npairs) {
    float2 p = src[i];
    dst[i] = pack2bf(p.x, p.y);
  }
}

// -------- fp32 W[k][cin][cout] -> bf16 Wt[k][cout][cin] (pair-packed) --------
__global__ void __launch_bounds__(256)
cast_w_transpose_kernel(const float* __restrict__ W,
                        unsigned int* __restrict__ Wt, int npairs) {
  int o = blockIdx.x * blockDim.x + threadIdx.x;   // pair index in Wt
  if (o >= npairs) return;
  int k     = o >> 11;            // / 2048 pairs per offset
  int rem   = o & 2047;
  int n     = rem >> 5;           // cout
  int cpair = rem & 31;           // cin pair
  const float* wk = W + (size_t)k * (CCH * CCH);
  float lo = wk[(size_t)(2 * cpair)     * CCH + n];
  float hi = wk[(size_t)(2 * cpair + 1) * CCH + n];
  Wt[o] = pack2bf(lo, hi);
}

// ---------------- submanifold conv via bf16 WMMA + fused BN stats ----------
// F    : [n][32] uints (bf16 pairs), Wt: [27][64][32] uints (Wt[cout][cin])
// nbr  : [27][n] int32, mask: [27][n] bool, zpad: 64B zeros
// Y    : [n][64] fp32, gstats: [128] fp32 (sum, sumsq) pre-zeroed
__global__ void __launch_bounds__(256)
subm_conv_wmma_kernel(const unsigned int* __restrict__ F,
                      const unsigned int* __restrict__ Wt,
                      const int* __restrict__ nbr,
                      const unsigned char* __restrict__ mask,
                      const unsigned int* __restrict__ zpad,
                      float* __restrict__ Y,
                      float* __restrict__ gstats, int n) {
  __shared__ __align__(16) unsigned int ldsA[2][TILE_M * 32]; // 2 x 128x64 bf16
  __shared__ __align__(16) unsigned int ldsW[2][CCH * 32];    // 2 x Wt tile
  __shared__ float sstat[128];

  const int tid  = threadIdx.x;
  const int lane = tid & 31;
  const int wv   = tid >> 5;
  const int base = blockIdx.x * TILE_M;
  const int half = lane >> 4;
  const int mrow = lane & 15;
  const int srow = tid >> 1;                 // staging row 0..127
  const int chlf = tid & 1;

  v8f acc[4];
#pragma unroll
  for (int t = 0; t < 4; ++t)
#pragma unroll
    for (int r = 0; r < 8; ++r) acc[t][r] = 0.0f;

  // ---- async-stage offset k into LDS buffer `buf` ----
  auto issue_stage = [&](int k, int buf) {
    int site = base + srow;
    const unsigned int* srcA = zpad;
    if (site < n) {
      size_t idx = (size_t)k * (size_t)n + (size_t)site;
      if (mask[idx]) srcA = F + (size_t)nbr[idx] * 32u + chlf * 16;
    }
    unsigned int aoff = (unsigned int)(size_t)&ldsA[buf][srow * 32 + chlf * 16];
#pragma unroll
    for (int j = 0; j < 4; ++j) async_ld128(aoff + j * 16, srcA + j * 4);
    const unsigned int* srcW = Wt + (size_t)k * (CCH * 32) + tid * 8;
    unsigned int woff = (unsigned int)(size_t)&ldsW[buf][tid * 8];
    async_ld128(woff, srcW);
    async_ld128(woff + 16, srcW + 4);
  };

  issue_stage(0, 0);

  for (int k = 0; k < KOFF; ++k) {
    const int cur = k & 1;
    wait_async0();              // this wave's staged loads complete (in order)
    __syncthreads();            // everyone's tile `cur` is resident
    if (k + 1 < KOFF) issue_stage(k + 1, cur ^ 1);  // overlap with WMMAs below

#pragma unroll
    for (int kc = 0; kc < 2; ++kc) {         // two 32-wide K chunks of Cin
      union { v16bf v; unsigned int u[8]; } a;
      const unsigned int* arow = &ldsA[cur][(wv * 16 + mrow) * 32 + kc * 16];
#pragma unroll
      for (int v = 0; v < 8; ++v) {
        int k0 = 2 * v + 16 * half; // A 16x32 bf16 layout, contiguous k = 16*half + s
        a.u[v] = arow[k0 >> 1];
      }
#pragma unroll
      for (int nt = 0; nt < 4; ++nt) {       // 4 x 16 output channels
        union { v16bf v; unsigned int u[8]; } b;
        const unsigned int* brow =
            &ldsW[cur][((lane & 15) + nt * 16) * 32 + kc * 16 + 8 * half];
#pragma unroll
        for (int v = 0; v < 8; ++v) b.u[v] = brow[v];
        acc[nt] = __builtin_amdgcn_wmma_f32_16x16x32_bf16(
            false, a.v, false, b.v, (short)0, acc[nt], false, false);
      }
    }
  }

  // ---- write D (vgpr r -> M = r + 8*half, N = lane&15) ----
#pragma unroll
  for (int nt = 0; nt < 4; ++nt) {
#pragma unroll
    for (int r = 0; r < 8; ++r) {
      int site = base + wv * 16 + r + 8 * half;
      if (site < n)
        Y[(size_t)site * CCH + nt * 16 + (lane & 15)] = acc[nt][r];
    }
  }

  // ---- fused BN partial statistics (tail rows are exactly zero) ----
  if (tid < 128) sstat[tid] = 0.0f;
  __syncthreads();
#pragma unroll
  for (int nt = 0; nt < 4; ++nt) {
    float s = 0.f, q = 0.f;
#pragma unroll
    for (int r = 0; r < 8; ++r) {
      float v = acc[nt][r];
      s += v;
      q += v * v;
    }
    int ch = nt * 16 + (lane & 15);
    atomicAdd(&sstat[ch], s);        // ds_add_f32
    atomicAdd(&sstat[64 + ch], q);
  }
  __syncthreads();
  if (tid < 128) atomicAdd(&gstats[tid], sstat[tid]);
}

// ---------------- BN finalize: per-channel scale/shift ----------------
__global__ void bn_finalize_kernel(const float* __restrict__ stats,
                                   const float* __restrict__ g,
                                   const float* __restrict__ beta,
                                   float* __restrict__ sc, int n) {
  int c = threadIdx.x;                       // 64 threads
  float invn  = 1.0f / (float)n;
  float mu    = stats[c] * invn;
  float var   = stats[64 + c] * invn - mu * mu;
  float rinv  = rsqrtf(var + BN_EPS);
  float scale = g[c] * rinv;
  sc[c]      = scale;
  sc[64 + c] = beta[c] - mu * scale;
}

// ---------------- BN + leaky -> bf16 pairs (conv2 input) ----------------
__global__ void __launch_bounds__(256)
bn_leaky_tobf16_kernel(const float2* __restrict__ Y,
                       const float* __restrict__ sc,
                       unsigned int* __restrict__ X, int npairs) {
  int i = blockIdx.x * blockDim.x + threadIdx.x;
  if (i >= npairs) return;
  int c0 = (2 * i) & 63;
  float2 p = Y[i];
  float a = p.x * sc[c0]     + sc[64 + c0];
  float b = p.y * sc[c0 + 1] + sc[64 + c0 + 1];
  a = a > 0.f ? a : a * LEAK;
  b = b > 0.f ? b : b * LEAK;
  X[i] = pack2bf(a, b);
}

// ---------------- final: BN + residual + leaky (fp32 out) ----------------
__global__ void __launch_bounds__(256)
bn_residual_leaky_kernel(const float* __restrict__ Y,
                         const float* __restrict__ sc,
                         const float* __restrict__ feat,
                         float* __restrict__ out, int total) {
  int i = blockIdx.x * blockDim.x + threadIdx.x;
  if (i >= total) return;
  int c = i & 63;
  float v = Y[i] * sc[c] + sc[64 + c] + feat[i];
  out[i] = v > 0.f ? v : v * LEAK;
}

extern "C" void kernel_launch(void* const* d_in, const int* in_sizes, int n_in,
                              void* d_out, int out_size, void* d_ws, size_t ws_size,
                              hipStream_t stream) {
  const float*         feat  = (const float*)d_in[0];
  const float*         W1    = (const float*)d_in[1];
  const float*         g1    = (const float*)d_in[2];
  const float*         be1   = (const float*)d_in[3];
  const float*         W2    = (const float*)d_in[4];
  const float*         g2    = (const float*)d_in[5];
  const float*         be2   = (const float*)d_in[6];
  const int*           nbr   = (const int*)d_in[7];
  const unsigned char* mask  = (const unsigned char*)d_in[8];
  float*               out   = (float*)d_out;

  const int n = in_sizes[0] / CCH;           // number of active sites

  // ---- workspace carve-out (256B aligned) ----
  char* ws = (char*)d_ws;
  size_t off = 0;
  auto take = [&](size_t bytes) -> void* {
    void* p = (void*)(ws + off);
    off = (off + bytes + 255) & ~(size_t)255;
    return p;
  };
  unsigned int* F16  = (unsigned int*)take((size_t)n * 32 * 4);
  float*        Y    = (float*)take((size_t)n * CCH * 4);
  unsigned int* W1t  = (unsigned int*)take((size_t)KOFF * CCH * CCH * 2);
  unsigned int* W2t  = (unsigned int*)take((size_t)KOFF * CCH * CCH * 2);
  float*        ctrl = (float*)take(1280);   // stats1|stats2|zero-page
  float*        sc1  = (float*)take(128 * 4);
  float*        sc2  = (float*)take(128 * 4);
  float*        stats1 = ctrl;
  float*        stats2 = ctrl + 128;
  unsigned int* zpad   = (unsigned int*)(ctrl + 256);  // 64B zeros

  hipMemsetAsync(ctrl, 0, 1280, stream);

  const int pairsF = n * (CCH / 2);
  const int pairsW = KOFF * CCH * CCH / 2;
  const int convBlocks = (n + TILE_M - 1) / TILE_M;
  const int total = n * CCH;

  cast_pairs_kernel<<<(pairsF + 255) / 256, 256, 0, stream>>>(
      (const float2*)feat, F16, pairsF);
  cast_w_transpose_kernel<<<(pairsW + 255) / 256, 256, 0, stream>>>(
      W1, W1t, pairsW);
  cast_w_transpose_kernel<<<(pairsW + 255) / 256, 256, 0, stream>>>(
      W2, W2t, pairsW);

  // conv1 (+fused stats) -> BN1 -> leaky -> bf16
  subm_conv_wmma_kernel<<<convBlocks, 256, 0, stream>>>(
      F16, W1t, nbr, mask, zpad, Y, stats1, n);
  bn_finalize_kernel<<<1, 64, 0, stream>>>(stats1, g1, be1, sc1, n);
  bn_leaky_tobf16_kernel<<<(pairsF + 255) / 256, 256, 0, stream>>>(
      (const float2*)Y, sc1, F16, pairsF);   // F16 now holds conv2 input

  // conv2 (+fused stats) -> BN2 -> +residual -> leaky
  subm_conv_wmma_kernel<<<convBlocks, 256, 0, stream>>>(
      F16, W2t, nbr, mask, zpad, Y, stats2, n);
  bn_finalize_kernel<<<1, 64, 0, stream>>>(stats2, g2, be2, sc2, n);
  bn_residual_leaky_kernel<<<(total + 255) / 256, 256, 0, stream>>>(
      Y, sc2, feat, out, total);
}
